// HungarianMatcher_49770081026747
// MI455X (gfx1250) — matrix operand order repaired and checked by reading, not compiled, not run on hardware
//
#include <hip/hip_runtime.h>
#include <hip/hip_bf16.h>

typedef __attribute__((ext_vector_type(16))) _Float16 v16h;
typedef __attribute__((ext_vector_type(8)))  float    v8f;

#define BS 8
#define NQ 200
#define NT 20
#define HW 25600
#define NC 80
#define FALPHA 0.25f
#define ROW_TILES 13              // ceil(200/16)
#define WAVES 8
#define KSLICE (HW / WAVES)       // 3200
#define KITERS (KSLICE / 32)      // 100

// ---------------- helper: load 16 consecutive f32, convert to packed f16 ----
static __device__ __forceinline__ v16h load16h(const float* __restrict__ p) {
    float4 r0 = *(const float4*)(p);
    float4 r1 = *(const float4*)(p + 4);
    float4 r2 = *(const float4*)(p + 8);
    float4 r3 = *(const float4*)(p + 12);
    v16h v;
    v[0]  = (_Float16)r0.x; v[1]  = (_Float16)r0.y; v[2]  = (_Float16)r0.z; v[3]  = (_Float16)r0.w;
    v[4]  = (_Float16)r1.x; v[5]  = (_Float16)r1.y; v[6]  = (_Float16)r1.z; v[7]  = (_Float16)r1.w;
    v[8]  = (_Float16)r2.x; v[9]  = (_Float16)r2.y; v[10] = (_Float16)r2.z; v[11] = (_Float16)r2.w;
    v[12] = (_Float16)r3.x; v[13] = (_Float16)r3.y; v[14] = (_Float16)r3.z; v[15] = (_Float16)r3.w;
    return v;
}

// ---------------- kernel A: t_sum[b][m] = sum_hw tgt_masks ------------------
__global__ void hm_tsum_kernel(const float* __restrict__ tgt_masks,
                               float* __restrict__ tsum) {
    const int idx = blockIdx.x;                 // b*NT + m
    const float* src = tgt_masks + (size_t)idx * HW;
    float s = 0.f;
    for (int i = threadIdx.x; i < HW; i += blockDim.x) s += src[i];
    __shared__ float red[256];
    red[threadIdx.x] = s;
    __syncthreads();
    for (int off = 128; off > 0; off >>= 1) {
        if ((int)threadIdx.x < off) red[threadIdx.x] += red[threadIdx.x + off];
        __syncthreads();
    }
    if (threadIdx.x == 0) tsum[idx] = red[0];
}

// ---------------- kernel B: per-(b,n) softmax stats: max, log-sum-exp -------
__global__ void hm_clsstat_kernel(const float* __restrict__ pred_classes,
                                  float* __restrict__ cmax,
                                  float* __restrict__ clse) {
    const int i = blockIdx.x * blockDim.x + threadIdx.x;
    if (i >= BS * NQ) return;
    const float* row = pred_classes + (size_t)i * NC;
    float m = -3.0e38f;
    for (int c = 0; c < NC; ++c) m = fmaxf(m, row[c]);
    float s = 0.f;
    for (int c = 0; c < NC; ++c) s += __expf(row[c] - m);
    cmax[i] = m;
    clse[i] = __logf(s);
}

// ---------------- kernel C: fused focal/dice WMMA cost matrix ---------------
__launch_bounds__(32 * WAVES)
__global__ void hm_cost_kernel(const float* __restrict__ pred_logits,
                               const float* __restrict__ pred_masks,
                               const float* __restrict__ pred_classes,
                               const float* __restrict__ tgt_masks,
                               const int*   __restrict__ tgt_classes,
                               const float* __restrict__ tsum,
                               const float* __restrict__ cmax,
                               const float* __restrict__ clse,
                               float* __restrict__ out) {
    const int b        = blockIdx.y;
    const int row_base = blockIdx.x * 16;
    const int lane     = threadIdx.x & 31;
    const int w        = threadIdx.x >> 5;
    const int hi       = lane >> 4;     // half-wave select
    const int lo       = lane & 15;

    // A-operand row for this lane (ISA 16-bit A 16x32 layout: lane%16 == M)
    int arow = row_base + lo; if (arow > NQ - 1) arow = NQ - 1;
    const float* aptr = pred_masks + (size_t)(b * NQ + arow) * HW;

    // B-operand columns (B 32x16: lane%16 == N, hi selects K half in this code's contiguous k order)
    const int m1ok = (16 + lo) < NT;
    const float* bptr0 = tgt_masks + (size_t)(b * NT + lo) * HW;
    const float* bptr1 = tgt_masks + (size_t)(b * NT + (m1ok ? (16 + lo) : 0)) * HW;

    v8f accM0 = {}, accM1 = {}, accP0 = {}, accP1 = {};
    float lsum_p = 0.f, lsum_fn = 0.f;

    const int kbase = w * KSLICE;
    for (int it = 0; it < KITERS; ++it) {
        const int k0 = kbase + it * 32;

        // ---- A: 16 pred_masks logits -> focal_diff (f16) and p (f16) ----
        const float* ap = aptr + k0 + hi * 16;
        float4 q0 = *(const float4*)(ap);
        float4 q1 = *(const float4*)(ap + 4);
        float4 q2 = *(const float4*)(ap + 8);
        float4 q3 = *(const float4*)(ap + 12);
        float xs[16] = {q0.x, q0.y, q0.z, q0.w, q1.x, q1.y, q1.z, q1.w,
                        q2.x, q2.y, q2.z, q2.w, q3.x, q3.y, q3.z, q3.w};
        if (it + 1 < KITERS) __builtin_prefetch(aptr + k0 + 32, 0, 1);

        v16h aD, aP;
        #pragma unroll
        for (int i = 0; i < 16; ++i) {
            float x   = xs[i];
            float e   = __expf(-x);
            float p   = 1.f / (1.f + e);
            float sp  = __logf(1.f + e);        // softplus(-x) = -log(p)
            float omp = e * p;                  // 1 - p
            float fpos = FALPHA * omp * omp * sp;
            float fneg = (1.f - FALPHA) * p * p * (x + sp);  // x+sp = softplus(x)
            lsum_p  += p;
            lsum_fn += fneg;
            aD[i] = (_Float16)(fpos - fneg);
            aP[i] = (_Float16)p;
        }

        // ---- B tiles: binary targets ----
        const int kb = k0 + hi * 16;
        v16h b0 = load16h(bptr0 + kb);
        v16h b1 = {};
        if (m1ok) b1 = load16h(bptr1 + kb);

        // ---- contraction on the matrix pipe ----
        accM0 = __builtin_amdgcn_wmma_f32_16x16x32_f16(false, aD, false, b0, (short)0, accM0, false, false);
        accP0 = __builtin_amdgcn_wmma_f32_16x16x32_f16(false, aP, false, b0, (short)0, accP0, false, false);
        accM1 = __builtin_amdgcn_wmma_f32_16x16x32_f16(false, aD, false, b1, (short)0, accM1, false, false);
        accP1 = __builtin_amdgcn_wmma_f32_16x16x32_f16(false, aP, false, b1, (short)0, accP1, false, false);
    }

    // ---- cross-wave reduction in LDS ----
    __shared__ float sAcc[WAVES][4][8][32];
    __shared__ float sP[WAVES][32];
    __shared__ float sF[WAVES][32];
    __shared__ float rowP[16];
    __shared__ float rowF[16];

    #pragma unroll
    for (int v = 0; v < 8; ++v) {
        sAcc[w][0][v][lane] = accM0[v];
        sAcc[w][1][v][lane] = accM1[v];
        sAcc[w][2][v][lane] = accP0[v];
        sAcc[w][3][v][lane] = accP1[v];
    }
    sP[w][lane] = lsum_p;
    sF[w][lane] = lsum_fn;
    __syncthreads();

    if (w == 0 && lo == lane) {       // lanes 0..15 of wave 0
        float rp = 0.f, rf = 0.f;
        #pragma unroll
        for (int ww = 0; ww < WAVES; ++ww) {
            rp += sP[ww][lane] + sP[ww][lane + 16];
            rf += sF[ww][lane] + sF[ww][lane + 16];
        }
        rowP[lane] = rp;
        rowF[lane] = rf;
    }
    __syncthreads();

    if (w == 0) {
        // D-matrix layout: VGPR v, lane L -> M = v + 8*(L>>4), N = L&15
        #pragma unroll
        for (int v = 0; v < 8; ++v) {
            float dM0 = 0.f, dM1 = 0.f, dP0 = 0.f, dP1 = 0.f;
            #pragma unroll
            for (int ww = 0; ww < WAVES; ++ww) {
                dM0 += sAcc[ww][0][v][lane];
                dM1 += sAcc[ww][1][v][lane];
                dP0 += sAcc[ww][2][v][lane];
                dP1 += sAcc[ww][3][v][lane];
            }
            const int M = v + 8 * hi;
            const int n = row_base + M;
            if (n >= NQ) continue;

            const float rp  = rowP[M];
            const float rf  = rowF[M];
            const float ccl = -pred_logits[(size_t)(b * NQ + n) * 2];
            const float cm  = cmax[b * NQ + n];
            const float cl  = clse[b * NQ + n];

            #pragma unroll
            for (int tile = 0; tile < 2; ++tile) {
                const int m = tile * 16 + lo;
                if (m >= NT) break;
                const float gm = tile ? dM1 : dM0;
                const float gd = tile ? dP1 : dP0;

                float mask_c = (gm + rf) * (1.f / HW);
                if (mask_c != mask_c) mask_c = 1e6f;

                float dice = 1.f - (2.f * gd + 1.f) / (rp + tsum[b * NT + m] + 1.f);
                if (dice != dice) dice = 1e6f;

                const int tc = tgt_classes[b * NT + m];
                const float logit = pred_classes[(size_t)(b * NQ + n) * NC + tc];
                const float pt  = __expf(logit - cm - cl);
                const float omt = 1.f - pt;
                const float clp = FALPHA * omt * omt * (-__logf(pt + 1e-8f));

                out[(size_t)(b * NQ + n) * NT + m] = mask_c + ccl + dice + clp;
            }
        }
    }
}

extern "C" void kernel_launch(void* const* d_in, const int* in_sizes, int n_in,
                              void* d_out, int out_size, void* d_ws, size_t ws_size,
                              hipStream_t stream) {
    const float* pred_logits  = (const float*)d_in[0];
    const float* pred_masks   = (const float*)d_in[1];
    const float* pred_classes = (const float*)d_in[2];
    const float* tgt_masks    = (const float*)d_in[3];
    const int*   tgt_classes  = (const int*)d_in[4];
    float* out = (float*)d_out;

    float* ws_tsum = (float*)d_ws;             // BS*NT   = 160
    float* ws_cmax = ws_tsum + BS * NT;        // BS*NQ   = 1600
    float* ws_clse = ws_cmax + BS * NQ;        // BS*NQ   = 1600

    hm_tsum_kernel<<<BS * NT, 256, 0, stream>>>(tgt_masks, ws_tsum);
    hm_clsstat_kernel<<<(BS * NQ + 255) / 256, 256, 0, stream>>>(pred_classes, ws_cmax, ws_clse);

    dim3 grid(ROW_TILES, BS);
    hm_cost_kernel<<<grid, 32 * WAVES, 0, stream>>>(pred_logits, pred_masks, pred_classes,
                                                    tgt_masks, tgt_classes,
                                                    ws_tsum, ws_cmax, ws_clse, out);
}
